// VVCrossAttention_7026566496882
// MI455X (gfx1250) — matrix operand order repaired and checked by reading, not compiled, not run on hardware
//
#include <hip/hip_runtime.h>
#include <hip/hip_bf16.h>

typedef __attribute__((ext_vector_type(16))) _Float16 v16h;
typedef __attribute__((ext_vector_type(8)))  _Float16 v8h;
typedef __attribute__((ext_vector_type(8)))  float    v8f;

#define C_DIM 512
#define BATCH 4
#define SEQ   4096
#define SCALE 0.125f

// ---------------- CDNA5 async global->LDS helpers (ASYNCcnt-tracked DMA) ---------------
__device__ __forceinline__ unsigned lds_off(const void* p) {
  // generic pointers to LDS are {aperture_hi32, lds_offset32}; low 32 bits = LDS offset
  return (unsigned)(unsigned long long)(uintptr_t)p;
}
__device__ __forceinline__ void async_ld_b128(unsigned loff, const _Float16* g) {
  asm volatile("global_load_async_to_lds_b128 %0, %1, off"
               :: "v"(loff), "v"((unsigned long long)(uintptr_t)g)
               : "memory");
}
__device__ __forceinline__ void wait_async_le8() {
  asm volatile("s_wait_asynccnt 8" ::: "memory");
}
__device__ __forceinline__ void wait_async_0() {
  asm volatile("s_wait_asynccnt 0" ::: "memory");
}

// ---------------- WMMA fragment builders (wave32, contiguous k order in both A and B) ----------------
// A: 16x32 f16. lanes 0-15: row=lane, halves[0:16]=K0..15
//               lanes16-31: row=lane-16, halves[0:16]=K16..31
__device__ __forceinline__ v16h frag_a_lds(const _Float16* base, int lda, int lane) {
  const int m  = lane & 15;
  const int k0 = (lane & 16) ? 16 : 0;
  const _Float16* p = base + m * lda + k0;
  union { v16h v; v8h h[2]; } u;
  u.h[0] = *(const v8h*)(p);
  u.h[1] = *(const v8h*)(p + 8);
  return u.v;
}

// B: 32x16 f16, column j's K-values contiguous.
// lanes 0-15: col=lane, K=0..15 ; lanes 16-31: col=lane-16, K=16..31
__device__ __forceinline__ v16h frag_b_ld(const _Float16* base, int ldcol, int lane) {
  const int j  = lane & 15;
  const int k0 = (lane & 16) ? 16 : 0;
  return *(const v16h*)(base + j * ldcol + k0);
}

// ---------------- Kernel 1: Y[r,d] = sum_c X[r,c]*W[d,c]  (fp32 in, f16 out) -----------
// Optionally also writes a per-batch transposed copy YT[b][d][m] (for P@V B-fragments).
__global__ __launch_bounds__(256) void proj_f32_to_f16(
    const float* __restrict__ X,   // [16384, 512]
    const float* __restrict__ W,   // [512, 512]
    _Float16* __restrict__ Y,      // [16384, 512] f16
    _Float16* __restrict__ YT)     // [4][512][4096] f16 or nullptr
{
  __shared__ _Float16 A_lds[64 * 32];
  __shared__ _Float16 B_lds[64 * 32];
  const int tid = threadIdx.x, lane = tid & 31, w = tid >> 5;
  const int wr = w & 3, wc = w >> 2;
  const int row0 = blockIdx.x * 64;
  const int col0 = blockIdx.y * 64;
  v8f acc0 = {}; v8f acc1 = {};
  for (int c0 = 0; c0 < C_DIM; c0 += 32) {
    const int idx = tid * 8;            // 2048 elements each for A and B
    const int r = idx >> 5, cc = idx & 31;
    const float* srca = X + (size_t)(row0 + r) * C_DIM + c0 + cc;
    const float* srcb = W + (size_t)(col0 + r) * C_DIM + c0 + cc;
    _Float16* dsta = A_lds + r * 32 + cc;
    _Float16* dstb = B_lds + r * 32 + cc;
#pragma unroll
    for (int i = 0; i < 8; ++i) { dsta[i] = (_Float16)srca[i]; dstb[i] = (_Float16)srcb[i]; }
    __syncthreads();
    v16h a  = frag_a_lds(A_lds + wr * 16 * 32, 32, lane);
    v16h b0 = frag_b_ld(B_lds + (wc * 32 +  0) * 32, 32, lane);
    v16h b1 = frag_b_ld(B_lds + (wc * 32 + 16) * 32, 32, lane);
    acc0 = __builtin_amdgcn_wmma_f32_16x16x32_f16(false, a, false, b0, (short)0, acc0, false, false);
    acc1 = __builtin_amdgcn_wmma_f32_16x16x32_f16(false, a, false, b1, (short)0, acc1, false, false);
    __syncthreads();
  }
  const int colL = lane & 15;
  const int rowH = (lane & 16) ? 8 : 0;
#pragma unroll
  for (int r = 0; r < 8; ++r) {
    const int row = row0 + wr * 16 + rowH + r;
    const int c   = col0 + wc * 32 + colL;
    const _Float16 h0 = (_Float16)acc0[r];
    const _Float16 h1 = (_Float16)acc1[r];
    Y[(size_t)row * C_DIM + c]      = h0;
    Y[(size_t)row * C_DIM + c + 16] = h1;
    if (YT) {
      const size_t bb = (size_t)(row >> 12);     // batch
      const int    m  = row & 4095;
      YT[(bb * C_DIM + c)      * SEQ + m] = h0;
      YT[(bb * C_DIM + c + 16) * SEQ + m] = h1;
    }
  }
}

// ---------------- Kernel 2: flash attention, post-softmax key masking, scrambled store -
__global__ __launch_bounds__(256) void flash_attn(
    const _Float16* __restrict__ Qh,   // [B,4096,512] f16
    const _Float16* __restrict__ Kh,   // [B,4096,512] f16 (K == V)
    const _Float16* __restrict__ KhT,  // [B,512,4096] f16 (transposed copy, L2-resident)
    const int* __restrict__ mask,      // [4096]
    _Float16* __restrict__ O2h)        // [B,4096,512] f16, scrambled layout
{
  __shared__ _Float16 Q_lds[64 * 512];        // 64 KB (resident Q tile)
  __shared__ _Float16 K_lds[2][32 * 512];     // 2 x 32 KB (double-buffered K tile)
  __shared__ float    S_lds[64 * 32];         // 8 KB
  __shared__ _Float16 P_lds[64 * 32];         // 4 KB
  __shared__ float    mrow[64], lrow[64], arow[64];
  __shared__ int      mk[32];

  const int tid = threadIdx.x, lane = tid & 31, w = tid >> 5;
  const int b  = blockIdx.x >> 6;
  const int n0 = (blockIdx.x & 63) * 64;
  const _Float16* Qb  = Qh  + (size_t)b * SEQ * C_DIM;
  const _Float16* Kb  = Kh  + (size_t)b * SEQ * C_DIM;
  const _Float16* KTb = KhT + (size_t)b * C_DIM * SEQ;

  // async DMA of resident Q tile (64x512 halves): 16 x b128 per wave
#pragma unroll
  for (int i = 0; i < 16; ++i) {
    const int idx = tid * 128 + i * 8;             // half index
    const int r = idx >> 9, cc = idx & 511;
    async_ld_b128(lds_off(Q_lds) + (unsigned)idx * 2,
                  Qb + (size_t)(n0 + r) * C_DIM + cc);
  }
  // async DMA of K tile 0: 8 x b128 per wave
#pragma unroll
  for (int i = 0; i < 8; ++i) {
    const int idx = tid * 64 + i * 8;
    const int kr = idx >> 9, cc = idx & 511;
    async_ld_b128(lds_off(K_lds[0]) + (unsigned)idx * 2,
                  Kb + (size_t)kr * C_DIM + cc);
  }
  if (tid < 64) { mrow[tid] = -__builtin_inff(); lrow[tid] = 0.f; }

  v8f acc[16];
#pragma unroll
  for (int t = 0; t < 16; ++t) acc[t] = (v8f){};

  const int Srow = (w & 3) * 16;
  const int Scol = (w >> 2) * 16;
  const int colbase = (w >> 2) * 256;
  const int NT = SEQ / 32;

  for (int kt = 0; kt < NT; ++kt) {
    const int m0  = kt * 32;
    const int cur = kt & 1;
    // prefetch next K tile into the other buffer, then wait for the current one
    if (kt + 1 < NT) {
      const _Float16* Kn = Kb + (size_t)(m0 + 32) * C_DIM;
#pragma unroll
      for (int i = 0; i < 8; ++i) {
        const int idx = tid * 64 + i * 8;
        const int kr = idx >> 9, cc = idx & 511;
        async_ld_b128(lds_off(K_lds[cur ^ 1]) + (unsigned)idx * 2,
                      Kn + (size_t)kr * C_DIM + cc);
      }
      wait_async_le8();   // in-order completion: current tile (older 8 ops) has landed
    } else {
      wait_async_0();
    }
    __syncthreads();

    if (tid < 32) mk[tid] = mask[m0 + tid];

    // S = Q * K^T (each wave: one 16x16 tile, 512-deep)
    v8f s = {};
    const _Float16* qbase = Q_lds + Srow * 512;
    const _Float16* kbase = K_lds[cur] + Scol * 512;
#pragma unroll
    for (int d0 = 0; d0 < 512; d0 += 32) {
      v16h a  = frag_a_lds(qbase + d0, 512, lane);
      v16h bf = frag_b_ld(kbase + d0, 512, lane);
      s = __builtin_amdgcn_wmma_f32_16x16x32_f16(false, a, false, bf, (short)0, s, false, false);
    }
    {
      const int colj  = Scol + (lane & 15);
      const int rbase = Srow + ((lane & 16) ? 8 : 0);
#pragma unroll
      for (int r = 0; r < 8; ++r) S_lds[(rbase + r) * 32 + colj] = s[r] * SCALE;
    }
    __syncthreads();

    // online softmax: full-row denominator, mask applied only to P used in P@V
    if (tid < 64) {
      const float mold = mrow[tid];
      float mx = mold;
      const float* srow = S_lds + tid * 32;
#pragma unroll
      for (int j = 0; j < 32; ++j) mx = fmaxf(mx, srow[j]);
      const float alpha = __expf(mold - mx);
      float sum = 0.f;
      _Float16* prow = P_lds + tid * 32;
#pragma unroll
      for (int j = 0; j < 32; ++j) {
        const float p = __expf(srow[j] - mx);
        sum += p;
        prow[j] = (_Float16)(mk[j] ? p : 0.f);
      }
      lrow[tid] = lrow[tid] * alpha + sum;
      mrow[tid] = mx;
      arow[tid] = alpha;
    }
    __syncthreads();

    // rescale acc, then acc += P * V   (V == K; B-fragments straight from L2 via KhT)
    {
      const int rbase = (w & 3) * 16 + ((lane & 16) ? 8 : 0);
      float al[8];
#pragma unroll
      for (int r = 0; r < 8; ++r) al[r] = arow[rbase + r];
#pragma unroll
      for (int t = 0; t < 16; ++t)
#pragma unroll
        for (int r = 0; r < 8; ++r) acc[t][r] = acc[t][r] * al[r];

      v16h pfr = frag_a_lds(P_lds + (w & 3) * 16 * 32, 32, lane);
      const int k0 = (lane & 16) ? 16 : 0;
#pragma unroll
      for (int t = 0; t < 16; ++t) {
        const int dcol = colbase + t * 16 + (lane & 15);
        v16h bv = *(const v16h*)(KTb + (size_t)dcol * SEQ + m0 + k0);
        acc[t] = __builtin_amdgcn_wmma_f32_16x16x32_f16(false, pfr, false, bv, (short)0, acc[t], false, false);
      }
    }
    __syncthreads();
  }

  // normalize and store in scrambled layout: O2[b, 8c + n>>9, n&511] = O[b,n,c]
  _Float16* Ob = O2h + (size_t)b * SEQ * C_DIM;
#pragma unroll
  for (int r = 0; r < 8; ++r) {
    const int rloc = (w & 3) * 16 + ((lane & 16) ? 8 : 0) + r;
    const float inv = 1.f / lrow[rloc];
    const int n = n0 + rloc;
#pragma unroll
    for (int t = 0; t < 16; ++t) {
      const int c = colbase + t * 16 + (lane & 15);
      Ob[(size_t)(8 * c + (n >> 9)) * C_DIM + (n & 511)] = (_Float16)(acc[t][r] * inv);
    }
  }
}

// ---------------- Kernel 3: out[r,d] = sum_c A[r,c]*Wp[d,c] + bp[d] (f16 in, f32 out) --
__global__ __launch_bounds__(256) void out_proj(
    const _Float16* __restrict__ A,   // [16384, 512] f16 (already scrambled)
    const float* __restrict__ W,      // Wp [512, 512]
    const float* __restrict__ bias,   // [512]
    float* __restrict__ Y)            // [16384, 512] f32
{
  __shared__ _Float16 A_lds[64 * 32];
  __shared__ _Float16 B_lds[64 * 32];
  const int tid = threadIdx.x, lane = tid & 31, w = tid >> 5;
  const int wr = w & 3, wc = w >> 2;
  const int row0 = blockIdx.x * 64;
  const int col0 = blockIdx.y * 64;
  v8f acc0 = {}; v8f acc1 = {};
  for (int c0 = 0; c0 < C_DIM; c0 += 32) {
    const int idx = tid * 8;
    const int r = idx >> 5, cc = idx & 31;
    // A tile via async DMA (1 b128 op per wave)
    async_ld_b128(lds_off(A_lds) + (unsigned)idx * 2,
                  A + (size_t)(row0 + r) * C_DIM + c0 + cc);
    const float* srcb = W + (size_t)(col0 + r) * C_DIM + c0 + cc;
    _Float16* dstb = B_lds + r * 32 + cc;
#pragma unroll
    for (int i = 0; i < 8; ++i) dstb[i] = (_Float16)srcb[i];
    wait_async_0();
    __syncthreads();
    v16h a  = frag_a_lds(A_lds + wr * 16 * 32, 32, lane);
    v16h b0 = frag_b_ld(B_lds + (wc * 32 +  0) * 32, 32, lane);
    v16h b1 = frag_b_ld(B_lds + (wc * 32 + 16) * 32, 32, lane);
    acc0 = __builtin_amdgcn_wmma_f32_16x16x32_f16(false, a, false, b0, (short)0, acc0, false, false);
    acc1 = __builtin_amdgcn_wmma_f32_16x16x32_f16(false, a, false, b1, (short)0, acc1, false, false);
    __syncthreads();
  }
  const int colL = lane & 15;
  const int rowH = (lane & 16) ? 8 : 0;
#pragma unroll
  for (int r = 0; r < 8; ++r) {
    const int row = row0 + wr * 16 + rowH + r;
    const int c   = col0 + wc * 32 + colL;
    Y[(size_t)row * C_DIM + c]      = acc0[r] + bias[c];
    Y[(size_t)row * C_DIM + c + 16] = acc1[r] + bias[c + 16];
  }
}

// ---------------- launch ----------------------------------------------------------------
extern "C" void kernel_launch(void* const* d_in, const int* in_sizes, int n_in,
                              void* d_out, int out_size, void* d_ws, size_t ws_size,
                              hipStream_t stream) {
  const float* x        = (const float*)d_in[0];
  const float* support  = (const float*)d_in[1];
  const int*   attnmask = (const int*)d_in[2];
  const float* Wv       = (const float*)d_in[3];
  const float* Wp       = (const float*)d_in[4];
  const float* bp       = (const float*)d_in[5];
  float* out = (float*)d_out;

  const size_t TEN = (size_t)BATCH * SEQ * C_DIM;   // 8,388,608 elements
  _Float16* Qh  = (_Float16*)d_ws;
  _Float16* Kh  = Qh  + TEN;
  _Float16* KhT = Kh  + TEN;
  _Float16* O2h = KhT + TEN;

  dim3 blk(256);
  dim3 gProj((BATCH * SEQ) / 64, C_DIM / 64);       // 256 x 8
  proj_f32_to_f16<<<gProj, blk, 0, stream>>>(x, Wv, Qh, (_Float16*)nullptr);
  proj_f32_to_f16<<<gProj, blk, 0, stream>>>(support, Wv, Kh, KhT);
  flash_attn<<<dim3(BATCH * (SEQ / 64)), blk, 0, stream>>>(Qh, Kh, KhT, attnmask, O2h);
  out_proj<<<gProj, blk, 0, stream>>>(O2h, Wp, bp, out);
}
